// GPT_Q_45346264711718
// MI455X (gfx1250) — compile-verified, matrix operand order checked by reading, not run on hardware
//
#include <hip/hip_runtime.h>
#include <hip/hip_bf16.h>
#include <cstdint>
#include <cstddef>

// ---------------- model constants ----------------
constexpr int NL = 12;      // layers
constexpr int NH = 12;      // heads
constexpr int NC = 768;     // channels
constexpr int NT = 1024;    // sequence
constexpr int NB = 8;       // batch
constexpr int NV = 50304;   // vocab
constexpr int NHD = 64;     // head dim
constexpr int NM = NB * NT; // 8192 rows of the activation matrix

// ---------------- vector types ----------------
typedef __attribute__((ext_vector_type(16))) _Float16 v16h;
typedef __attribute__((ext_vector_type(8)))  _Float16 v8h;
typedef __attribute__((ext_vector_type(8)))  float    v8f;
typedef __attribute__((ext_vector_type(8)))  signed char c8;
typedef __attribute__((ext_vector_type(4)))  unsigned int u32x4;
typedef __attribute__((ext_vector_type(8)))  int i32x8;
typedef __attribute__((ext_vector_type(4)))  int i32x4;

// WMMA helper: D = A*B + C  (f16 inputs, f32 accum, 16x16x32)
__device__ __forceinline__ v8f wmma_f32_f16(v16h a, v16h b, v8f c) {
  return __builtin_amdgcn_wmma_f32_16x16x32_f16(false, a, false, b, (short)0, c,
                                                false, false);
}

// Load a 16x32 f16 A/B fragment row for this lane from a (16B-aligned) row
// pointer, following the CDNA5 ISA layout:
//   lane L (lr = L&15, kh = L>>4): elements 0..7  <-> K =      8*kh + 0..7
//                                  elements 8..15 <-> K = 16 + 8*kh + 0..7
__device__ __forceinline__ v16h load_frag(const _Float16* rowp, int kh) {
  v8h lo = *(const v8h*)(rowp + 8 * kh);
  v8h hi = *(const v8h*)(rowp + 16 + 8 * kh);
  return __builtin_shufflevector(lo, hi, 0, 1, 2, 3, 4, 5, 6, 7, 8, 9, 10, 11,
                                 12, 13, 14, 15);
}

// LDS byte offset of a __shared__ object (generic LDS address low 32 bits are
// the wave-relative LDS offset per ISA flat-addressing rules).
__device__ __forceinline__ unsigned lds_off(const void* p) {
  return (unsigned)(unsigned long long)(uintptr_t)p;
}

// ---------------- Tensor Data Mover: 2D f16 tile -> LDS ----------------
// D# per CDNA5 ISA:
//  group0: [1:0]=count=1, [63:32]=lds_addr, [120:64]=global_addr, [127:126]=type=2
//  group1: [17:16]=data_size(1=2B), [20]=pad_enable, [24:22]=pad_interval,
//          [31:25]=pad_amount, [79:48]=tensor_dim0, [111:80]=tensor_dim1,
//          [127:112]=tile_dim0, [143:128]=tile_dim1, [207:160]=tensor_dim0_stride
//  groups 2/3 (+trailing group): zero (2D tile, tile_dim2=0 unused)
// pad codes: interval 0=2,1=4,2=8,3=16,4=32.. DWORDs; amount N -> N+1 DWORDs
__device__ __forceinline__ void tdm_load_2d_f16(unsigned lds_addr,
                                                const void* gaddr,
                                                unsigned tensor_w,
                                                unsigned tensor_h,
                                                unsigned stride_elems,
                                                unsigned tile_w, unsigned tile_h,
                                                unsigned pad_int,
                                                unsigned pad_amt) {
  const unsigned long long ga = (unsigned long long)(uintptr_t)gaddr;
  u32x4 g0;
  g0.x = 1u;                                                   // count=1
  g0.y = lds_addr;
  g0.z = (unsigned)ga;                                         // addr[31:0]
  g0.w = (unsigned)((ga >> 32) & 0x01FFFFFFu) | (2u << 30);    // addr[56:32]|type=2
  i32x8 g1;
  g1[0] = (int)((1u << 16) | (1u << 20) | (pad_int << 22) | (pad_amt << 25));
  g1[1] = (int)((tensor_w & 0xFFFFu) << 16);                   // tensor_dim0 lo
  g1[2] = (int)((tensor_w >> 16) | ((tensor_h & 0xFFFFu) << 16));
  g1[3] = (int)((tensor_h >> 16) | (tile_w << 16));            // tile_dim0
  g1[4] = (int)tile_h;                                         // tile_dim1
  g1[5] = (int)stride_elems;                                   // dim0 stride lo32
  g1[6] = 0;
  g1[7] = 0;
  const i32x4 gz4 = {0, 0, 0, 0};
  const i32x8 gz8 = {0, 0, 0, 0, 0, 0, 0, 0};
  // 6-arg (clang-23 / therock) form: (g0, g1, g2, g3, g4, cpol)
  __builtin_amdgcn_tensor_load_to_lds(g0, g1, gz4, gz4, gz8, 0);
}

// ---------------- embedding: x = wte[idx] + wpe ----------------
__global__ __launch_bounds__(256)
void embed_kernel(const int* __restrict__ idx, const float* __restrict__ wte,
                  const float* __restrict__ wpe, float* __restrict__ x) {
  const int row = blockIdx.x;              // b*T + t
  const int t   = row % NT;
  const int tok = idx[row];
  const float* wt = wte + (size_t)tok * NC;
  const float* wp = wpe + (size_t)t * NC;
  float* xo = x + (size_t)row * NC;
  for (int c = threadIdx.x; c < NC; c += 256) xo[c] = wt[c] + wp[c];
}

// ---------------- dequant int8 vector -> f32 ----------------
__global__ __launch_bounds__(256)
void dequant_kernel(const signed char* __restrict__ q,
                    const float* __restrict__ scales, const int* __restrict__ zps,
                    int sidx, float* __restrict__ out, int n) {
  const int i = blockIdx.x * 256 + threadIdx.x;
  if (i < n) out[i] = scales[sidx] * (float)((int)q[i] - zps[sidx]);
}

// ---------------- layernorm (f32 in -> f16 out) ----------------
__global__ __launch_bounds__(256)
void ln_f16_kernel(const float* __restrict__ x, const float* __restrict__ w,
                   const float* __restrict__ bvec, _Float16* __restrict__ out) {
  __shared__ float rs[8], rs2[8], stat[2];
  const int row = blockIdx.x;
  const float* xp = x + (size_t)row * NC;
  float s = 0.f, s2 = 0.f;
  for (int c = threadIdx.x; c < NC; c += 256) {
    float v = xp[c];
    s += v; s2 += v * v;
  }
#pragma unroll
  for (int off = 16; off > 0; off >>= 1) {
    s  += __shfl_xor(s, off, 32);
    s2 += __shfl_xor(s2, off, 32);
  }
  if ((threadIdx.x & 31) == 0) { rs[threadIdx.x >> 5] = s; rs2[threadIdx.x >> 5] = s2; }
  __syncthreads();
  if (threadIdx.x == 0) {
    float ts = 0.f, ts2 = 0.f;
    for (int i = 0; i < 8; ++i) { ts += rs[i]; ts2 += rs2[i]; }
    const float mu  = ts / NC;
    const float var = ts2 / NC - mu * mu;
    stat[0] = mu; stat[1] = rsqrtf(var + 1e-5f);
  }
  __syncthreads();
  const float mu = stat[0], rstd = stat[1];
  _Float16* op = out + (size_t)row * NC;
  for (int c = threadIdx.x; c < NC; c += 256)
    op[c] = (_Float16)((xp[c] - mu) * rstd * w[c] + bvec[c]);
}

// ---------------- final layernorm on last positions only (f32 out) ----------------
__global__ __launch_bounds__(256)
void lnf_kernel(const float* __restrict__ x, const float* __restrict__ w,
                const float* __restrict__ bvec, float* __restrict__ out) {
  __shared__ float rs[8], rs2[8], stat[2];
  const int bi  = blockIdx.x;
  const int row = bi * NT + (NT - 1);
  const float* xp = x + (size_t)row * NC;
  float s = 0.f, s2 = 0.f;
  for (int c = threadIdx.x; c < NC; c += 256) {
    float v = xp[c];
    s += v; s2 += v * v;
  }
#pragma unroll
  for (int off = 16; off > 0; off >>= 1) {
    s  += __shfl_xor(s, off, 32);
    s2 += __shfl_xor(s2, off, 32);
  }
  if ((threadIdx.x & 31) == 0) { rs[threadIdx.x >> 5] = s; rs2[threadIdx.x >> 5] = s2; }
  __syncthreads();
  if (threadIdx.x == 0) {
    float ts = 0.f, ts2 = 0.f;
    for (int i = 0; i < 8; ++i) { ts += rs[i]; ts2 += rs2[i]; }
    const float mu  = ts / NC;
    const float var = ts2 / NC - mu * mu;
    stat[0] = mu; stat[1] = rsqrtf(var + 1e-5f);
  }
  __syncthreads();
  const float mu = stat[0], rstd = stat[1];
  for (int c = threadIdx.x; c < NC; c += 256)
    out[(size_t)bi * NC + c] = (xp[c] - mu) * rstd * w[c] + bvec[c];
}

// ---------------- WMMA GEMM with int8-weight dequant ----------------
// out[m,n] = act(sum_k A[m,k] * ws*(Wq[n,k]-wz) + bias[n])
// mode 0: store f16   mode 1: gelu -> f16   mode 2: resid[m,n] += val (f32)
// Block: 256 thr (8 waves), tile 128x64, K-step 32, double-buffered LDS.
// A tile is moved by the Tensor Data Mover (wave 0 issues, TENSORcnt waits);
// the TDM pad feature (16 DWORDs data + 4 DWORDs pad) generates the 40-half
// padded LDS rows directly. B tile goes through registers for int8->f16.
__global__ __launch_bounds__(256)
void gemm_i8w_kernel(const _Float16* __restrict__ A,
                     const signed char* __restrict__ Wq,
                     const float* __restrict__ bias,
                     const float* __restrict__ scales,
                     const int* __restrict__ zps, int sidx,
                     _Float16* __restrict__ outH, float* __restrict__ resid,
                     int Ndim, int Kdim, int mode) {
  __shared__ __align__(16) _Float16 lA[2][128][40];
  __shared__ __align__(16) _Float16 lB[2][64][40];

  const float wsc = scales[sidx];
  const int   wzp = zps[sidx];
  const int tid  = threadIdx.x;
  const int row0 = blockIdx.y * 128;
  const int col0 = blockIdx.x * 64;

  // B staging: 64x32 int8 tile, 8 bytes per thread
  const int bRow = tid >> 2;
  const int bK   = (tid & 3) * 8;
  const signed char* bPtr = Wq + (size_t)(col0 + bRow) * Kdim + bK;
  const _Float16*    aTile = A + (size_t)row0 * Kdim;

  const int wave = tid >> 5;
  const int lane = tid & 31;
  const int wm0  = (wave >> 1) * 32;  // 4x2 wave grid, 32x32 per wave
  const int wn0  = (wave & 1) * 32;
  const int lr   = lane & 15;
  const int kh   = lane >> 4;

  // uniform wave id so the TDM issue is a true scalar branch (TDM ignores EXEC)
  const unsigned wid = ((unsigned)__builtin_amdgcn_readfirstlane((int)threadIdx.x)) >> 5;
  const unsigned ldsA0 = lds_off(&lA[0][0][0]);
  constexpr unsigned ldsAbuf = 128u * 40u * 2u;   // bytes per A buffer

  const v8f zero8 = {0.f, 0.f, 0.f, 0.f, 0.f, 0.f, 0.f, 0.f};
  v8f acc[2][2];
  acc[0][0] = zero8; acc[0][1] = zero8; acc[1][0] = zero8; acc[1][1] = zero8;

  c8 rb;

  // prologue: stage k-block 0 (A via TDM, B via registers)
  if (wid == 0)
    tdm_load_2d_f16(ldsA0, aTile, (unsigned)Kdim, 128u, (unsigned)Kdim,
                    32u, 128u, 3u /*16 DW*/, 3u /*4 DW pad*/);
  rb = *(const c8*)(bPtr);
#pragma unroll
  for (int i = 0; i < 8; ++i)
    lB[0][bRow][bK + i] = (_Float16)(wsc * (float)((int)rb[i] - wzp));
  __builtin_amdgcn_s_wait_tensorcnt(0);
  __syncthreads();

  const int nk = Kdim >> 5;
  for (int kb = 0; kb < nk; ++kb) {
    const int buf  = kb & 1;
    const bool more = (kb + 1) < nk;
    if (more) {
      const int knext = (kb + 1) << 5;
      if (wid == 0)
        tdm_load_2d_f16(ldsA0 + (unsigned)(buf ^ 1) * ldsAbuf, aTile + knext,
                        (unsigned)Kdim, 128u, (unsigned)Kdim, 32u, 128u, 3u, 3u);
      __builtin_prefetch(bPtr + knext + 32, 0, 0);   // global_prefetch_b8
      rb = *(const c8*)(bPtr + knext);
    }
    v16h af[2], bf[2];
#pragma unroll
    for (int tm = 0; tm < 2; ++tm)
      af[tm] = load_frag(&lA[buf][wm0 + tm * 16 + lr][0], kh);
#pragma unroll
    for (int tn = 0; tn < 2; ++tn)
      bf[tn] = load_frag(&lB[buf][wn0 + tn * 16 + lr][0], kh);
#pragma unroll
    for (int tm = 0; tm < 2; ++tm)
#pragma unroll
      for (int tn = 0; tn < 2; ++tn)
        acc[tm][tn] = wmma_f32_f16(af[tm], bf[tn], acc[tm][tn]);
    if (more) {
      const int nb = buf ^ 1;
#pragma unroll
      for (int i = 0; i < 8; ++i)
        lB[nb][bRow][bK + i] = (_Float16)(wsc * (float)((int)rb[i] - wzp));
    }
    __builtin_amdgcn_s_wait_tensorcnt(0);
    __syncthreads();
  }

  // epilogue: C layout — M = vgpr + 8*kh, N = lr
#pragma unroll
  for (int tm = 0; tm < 2; ++tm)
#pragma unroll
    for (int tn = 0; tn < 2; ++tn) {
      const int n  = col0 + wn0 + tn * 16 + lr;
      const float bn = bias[n];
#pragma unroll
      for (int v = 0; v < 8; ++v) {
        const int m = row0 + wm0 + tm * 16 + v + 8 * kh;
        float val = acc[tm][tn][v] + bn;
        const size_t oi = (size_t)m * Ndim + n;
        if (mode == 0) {
          outH[oi] = (_Float16)val;
        } else if (mode == 1) {
          val = 0.5f * val * (1.0f + erff(val * 0.70710678118654752f));
          outH[oi] = (_Float16)val;
        } else {
          resid[oi] += val;
        }
      }
    }
}

// ---------------- flash attention (1 wave per 16 query rows) ----------------
// qkv: [B*T, 3C] f16 (q | k | v, head-major inside each C block)
// y  : [B*T, C]  f16
// V tiles (32 keys x 64 ch, row stride 3C) are copied by the TDM at the top of
// each key-block iteration; the TENSORcnt wait lands after Q.K^T + softmax.
__global__ __launch_bounds__(32)
void attn_kernel(const _Float16* __restrict__ qkv, _Float16* __restrict__ y) {
  __shared__ __align__(16) float    sS[16][32];
  __shared__ __align__(16) _Float16 sP[16][32];
  __shared__ __align__(16) _Float16 sV[32][72];
  __shared__ float sMax[16], sSum[16], sAlpha[16];

  const int qb   = blockIdx.x;   // T/16
  const int h    = blockIdx.y;   // heads
  const int b    = blockIdx.z;   // batch
  const int lane = threadIdx.x;
  const int lr   = lane & 15;
  const int kh   = lane >> 4;
  const int qrow0 = qb * 16;
  const int RS = 3 * NC;

  const _Float16* qbase = qkv + ((size_t)b * NT) * RS + h * NHD;
  const _Float16* kbase = qbase + NC;
  const _Float16* vbase = qbase + 2 * NC;
  const unsigned ldsV0 = lds_off(&sV[0][0]);

  // Q fragments (rows qrow0..+15, K = channel 0..63), pre-scaled by 1/sqrt(HD)
  v16h aq[2];
#pragma unroll
  for (int j = 0; j < 2; ++j) {
    v16h tq = load_frag(qbase + (size_t)(qrow0 + lr) * RS + j * 32, kh);
#pragma unroll
    for (int e = 0; e < 16; ++e) tq[e] = tq[e] * (_Float16)0.125f;
    aq[j] = tq;
  }

  const v8f zero8 = {0.f, 0.f, 0.f, 0.f, 0.f, 0.f, 0.f, 0.f};
  v8f o[4];
#pragma unroll
  for (int i = 0; i < 4; ++i) o[i] = zero8;
  if (lane < 16) { sMax[lane] = -1e30f; sSum[lane] = 0.0f; }

  const int nkb = (qrow0 + 47) >> 5;   // 32-key blocks covering causal range
  for (int kb = 0; kb < nkb; ++kb) {
    const int key0 = kb << 5;
    // async: TDM copies V tile [32 keys x 64 ch] into sV (72-half padded rows:
    // 32 DWORDs data + 4 DWORDs pad per row)
    tdm_load_2d_f16(ldsV0, vbase + (size_t)key0 * RS, 64u, 32u, (unsigned)RS,
                    64u, 32u, 4u /*32 DW*/, 3u /*4 DW pad*/);
    // S = Q @ K^T for two 16-key sub-blocks, masked, into LDS
#pragma unroll
    for (int sub = 0; sub < 2; ++sub) {
      const int n0 = key0 + sub * 16;
      v16h bk0 = load_frag(kbase + (size_t)(n0 + lr) * RS, kh);
      v16h bk1 = load_frag(kbase + (size_t)(n0 + lr) * RS + 32, kh);
      v8f s = zero8;
      s = wmma_f32_f16(aq[0], bk0, s);
      s = wmma_f32_f16(aq[1], bk1, s);
#pragma unroll
      for (int v = 0; v < 8; ++v) {
        const int m = v + 8 * kh;
        sS[m][sub * 16 + lr] = ((n0 + lr) <= (qrow0 + m)) ? s[v] : -1e30f;
      }
    }
    __syncthreads();
    // online softmax: lanes 0..15 own one query row each
    if (lane < 16) {
      const float mold = sMax[lane];
      float mx = mold;
#pragma unroll
      for (int j = 0; j < 32; ++j) mx = fmaxf(mx, sS[lane][j]);
      const float alpha = __expf(mold - mx);
      float sum = sSum[lane] * alpha;
#pragma unroll
      for (int j = 0; j < 32; ++j) {
        const float p = __expf(sS[lane][j] - mx);
        sP[lane][j] = (_Float16)p;
        sum += p;
      }
      sMax[lane] = mx; sSum[lane] = sum; sAlpha[lane] = alpha;
    }
    __syncthreads();
    // rescale accumulated O, then O += P @ V
#pragma unroll
    for (int ti = 0; ti < 4; ++ti)
#pragma unroll
      for (int v = 0; v < 8; ++v) o[ti][v] *= sAlpha[v + 8 * kh];
    v16h ap = load_frag(&sP[lr][0], kh);
    __builtin_amdgcn_s_wait_tensorcnt(0);   // V tile landed in sV
#pragma unroll
    for (int tn = 0; tn < 4; ++tn) {
      v16h bv;
#pragma unroll
      for (int e = 0; e < 8; ++e) {
        bv[e]     = sV[8 * kh + e][tn * 16 + lr];
        bv[8 + e] = sV[16 + 8 * kh + e][tn * 16 + lr];
      }
      o[tn] = wmma_f32_f16(ap, bv, o[tn]);
    }
    __syncthreads();
  }

  // write normalized output
#pragma unroll
  for (int tn = 0; tn < 4; ++tn)
#pragma unroll
    for (int v = 0; v < 8; ++v) {
      const int m = v + 8 * kh;
      const float val = o[tn][v] / sSum[m];
      y[((size_t)b * NT + qrow0 + m) * NC + h * NHD + tn * 16 + lr] =
          (_Float16)val;
    }
}

// ---------------- LM head: out[b,v] = dot(xf[b,:], wte[v,:]) ----------------
__global__ __launch_bounds__(256)
void lmhead_kernel(const float* __restrict__ xf, const float* __restrict__ wte,
                   float* __restrict__ out) {
  const int wave = threadIdx.x >> 5;
  const int lane = threadIdx.x & 31;
  const int v = blockIdx.x * 8 + wave;   // grid = V/8
  const float* wrow = wte + (size_t)v * NC;
  float acc[NB];
#pragma unroll
  for (int bb = 0; bb < NB; ++bb) acc[bb] = 0.f;
  for (int c = lane; c < NC; c += 32) {
    const float wv = wrow[c];
#pragma unroll
    for (int bb = 0; bb < NB; ++bb) acc[bb] += wv * xf[bb * NC + c];
  }
#pragma unroll
  for (int off = 16; off > 0; off >>= 1)
#pragma unroll
    for (int bb = 0; bb < NB; ++bb) acc[bb] += __shfl_xor(acc[bb], off, 32);
  if (lane == 0)
#pragma unroll
    for (int bb = 0; bb < NB; ++bb) out[(size_t)bb * NV + v] = acc[bb];
}

// ---------------- host ----------------
extern "C" void kernel_launch(void* const* d_in, const int* in_sizes, int n_in,
                              void* d_out, int out_size, void* d_ws,
                              size_t ws_size, hipStream_t stream) {
  (void)in_sizes; (void)n_in; (void)out_size; (void)ws_size;
  const signed char* ln1w_q  = (const signed char*)d_in[0];
  const signed char* ln1b_q  = (const signed char*)d_in[1];
  const signed char* attnw_q = (const signed char*)d_in[2];
  const signed char* attnb_q = (const signed char*)d_in[3];
  const signed char* attpw_q = (const signed char*)d_in[4];
  const signed char* attpb_q = (const signed char*)d_in[5];
  const signed char* ln2w_q  = (const signed char*)d_in[6];
  const signed char* ln2b_q  = (const signed char*)d_in[7];
  const signed char* fcw_q   = (const signed char*)d_in[8];
  const signed char* fcb_q   = (const signed char*)d_in[9];
  const signed char* mlpw_q  = (const signed char*)d_in[10];
  const signed char* mlpb_q  = (const signed char*)d_in[11];
  const int*   idx   = (const int*)d_in[12];
  const float* wte   = (const float*)d_in[13];
  const float* wpe   = (const float*)d_in[14];
  const float* lnfw  = (const float*)d_in[15];
  const float* lnfb  = (const float*)d_in[16];
  const float* scP   = (const float*)d_in[17];
  const int*   zpP   = (const int*)d_in[18];
  float* out = (float*)d_out;

  // workspace partition (256B aligned)
  char* ws = (char*)d_ws;
  size_t off = 0;
  auto wsa = [&](size_t bytes) {
    void* p = ws + off;
    off = (off + bytes + 255) & ~(size_t)255;
    return p;
  };
  float*    x    = (float*)   wsa((size_t)NM * NC * 4);      // residual stream
  _Float16* hbuf = (_Float16*)wsa((size_t)NM * NC * 2);      // LN output (GEMM A)
  _Float16* qkvb = (_Float16*)wsa((size_t)NM * 3 * NC * 2);  // qkv
  _Float16* ybuf = (_Float16*)wsa((size_t)NM * NC * 2);      // attention out
  _Float16* hmlp = (_Float16*)wsa((size_t)NM * 4 * NC * 2);  // gelu(fc) out
  float* v_ln1w  = (float*)wsa(NC * 4);
  float* v_ln1b  = (float*)wsa(NC * 4);
  float* v_attnb = (float*)wsa(3 * NC * 4);
  float* v_attpb = (float*)wsa(NC * 4);
  float* v_ln2w  = (float*)wsa(NC * 4);
  float* v_ln2b  = (float*)wsa(NC * 4);
  float* v_fcb   = (float*)wsa(4 * NC * 4);
  float* v_mlpb  = (float*)wsa(NC * 4);
  float* xf      = (float*)wsa((size_t)NB * NC * 4);

  embed_kernel<<<NM, 256, 0, stream>>>(idx, wte, wpe, x);

  for (int l = 0; l < NL; ++l) {
    const int si = l * 12;
    dequant_kernel<<<(NC + 255) / 256, 256, 0, stream>>>(ln1w_q + (size_t)l * NC, scP, zpP, si + 0, v_ln1w, NC);
    dequant_kernel<<<(NC + 255) / 256, 256, 0, stream>>>(ln1b_q + (size_t)l * NC, scP, zpP, si + 1, v_ln1b, NC);
    dequant_kernel<<<(3 * NC + 255) / 256, 256, 0, stream>>>(attnb_q + (size_t)l * 3 * NC, scP, zpP, si + 3, v_attnb, 3 * NC);
    dequant_kernel<<<(NC + 255) / 256, 256, 0, stream>>>(attpb_q + (size_t)l * NC, scP, zpP, si + 5, v_attpb, NC);
    dequant_kernel<<<(NC + 255) / 256, 256, 0, stream>>>(ln2w_q + (size_t)l * NC, scP, zpP, si + 6, v_ln2w, NC);
    dequant_kernel<<<(NC + 255) / 256, 256, 0, stream>>>(ln2b_q + (size_t)l * NC, scP, zpP, si + 7, v_ln2b, NC);
    dequant_kernel<<<(4 * NC + 255) / 256, 256, 0, stream>>>(fcb_q + (size_t)l * 4 * NC, scP, zpP, si + 9, v_fcb, 4 * NC);
    dequant_kernel<<<(NC + 255) / 256, 256, 0, stream>>>(mlpb_q + (size_t)l * NC, scP, zpP, si + 11, v_mlpb, NC);

    // ---- attention block ----
    ln_f16_kernel<<<NM, 256, 0, stream>>>(x, v_ln1w, v_ln1b, hbuf);
    gemm_i8w_kernel<<<dim3(3 * NC / 64, NM / 128), 256, 0, stream>>>(
        hbuf, attnw_q + (size_t)l * 3 * NC * NC, v_attnb, scP, zpP, si + 2,
        qkvb, (float*)nullptr, 3 * NC, NC, 0);
    attn_kernel<<<dim3(NT / 16, NH, NB), 32, 0, stream>>>(qkvb, ybuf);
    gemm_i8w_kernel<<<dim3(NC / 64, NM / 128), 256, 0, stream>>>(
        ybuf, attpw_q + (size_t)l * NC * NC, v_attpb, scP, zpP, si + 4,
        (_Float16*)nullptr, x, NC, NC, 2);

    // ---- mlp block ----
    ln_f16_kernel<<<NM, 256, 0, stream>>>(x, v_ln2w, v_ln2b, hbuf);
    gemm_i8w_kernel<<<dim3(4 * NC / 64, NM / 128), 256, 0, stream>>>(
        hbuf, fcw_q + (size_t)l * 4 * NC * NC, v_fcb, scP, zpP, si + 8,
        hmlp, (float*)nullptr, 4 * NC, NC, 1);
    gemm_i8w_kernel<<<dim3(NC / 64, NM / 128), 256, 0, stream>>>(
        hmlp, mlpw_q + (size_t)l * NC * 4 * NC, v_mlpb, scP, zpP, si + 10,
        (_Float16*)nullptr, x, NC, 4 * NC, 2);
  }

  lnf_kernel<<<NB, 256, 0, stream>>>(x, lnfw, lnfb, xf);
  lmhead_kernel<<<NV / 8, 256, 0, stream>>>(xf, wte, out);
}
